// GraphEmbeddingModule_28475633172511
// MI455X (gfx1250) — matrix operand order repaired and checked by reading, not compiled, not run on hardware
//
#include <hip/hip_runtime.h>

typedef __attribute__((ext_vector_type(16))) __bf16 v16bf;
typedef __attribute__((ext_vector_type(8)))  __bf16 v8bf;
typedef __attribute__((ext_vector_type(4)))  __bf16 v4bf;
typedef __attribute__((ext_vector_type(8)))  float  v8f;

union AFrag { v16bf v; uint4 q[2]; };

constexpr int FANOUT_C     = 16;
constexpr int NODES_PER_WG = 32;
constexpr int THREADS      = 128;

__global__ __launch_bounds__(256) void f32_to_bf16_kernel(
    const float* __restrict__ s, __bf16* __restrict__ d, int n) {
  int i = blockIdx.x * 256 + threadIdx.x;
  if (i < n) d[i] = (__bf16)s[i];
}

// One fused GraphSAGE layer:
//   A = [h_self | mean(h_neigh)]  (bf16, staged in LDS)
//   out = relu(A @ W.T)           (WMMA bf16 -> f32 accumulate)
// DIN = input feature dim, DOUT = output dim (multiple of 32).
template <int DIN, int DOUT, bool IN_F32, bool OUT_F32>
__global__ __launch_bounds__(THREADS) void sage_layer_kernel(
    const void* __restrict__ h_in_v, const int* __restrict__ neigh,
    const __bf16* __restrict__ Wbf, void* __restrict__ h_out_v) {
  constexpr int K2 = 2 * DIN;          // combined width
  constexpr int RS = K2 * 2 + 16;      // LDS row stride (bytes); +16 => 4-bank skew, conflict-free
  __shared__ __align__(16) unsigned char A_sh[NODES_PER_WG * RS];

  const int tid       = threadIdx.x;
  const int localNode = tid >> 2;      // 4 threads per node
  const int sub       = tid & 3;
  const int node      = blockIdx.x * NODES_PER_WG + localNode;

  // ---- Phase 1: gather + mean + concat into LDS (bf16) ----
  int nb[FANOUT_C];
  {
    const int* np = neigh + node * FANOUT_C;
#pragma unroll
    for (int f = 0; f < FANOUT_C; ++f) nb[f] = np[f];
  }
  __bf16* row = (__bf16*)(&A_sh[localNode * RS]);

  if constexpr (IN_F32) {
    const float* h_in  = (const float*)h_in_v;
    const float* selfp = h_in + (size_t)node * DIN;
    for (int c = sub * 4; c < DIN; c += 16) {
      float4 s = *(const float4*)(selfp + c);
      float ax = 0.f, ay = 0.f, az = 0.f, aw = 0.f;
#pragma unroll
      for (int f = 0; f < FANOUT_C; ++f) {
        float4 t = *(const float4*)(h_in + (size_t)nb[f] * DIN + c);
        ax += t.x; ay += t.y; az += t.z; aw += t.w;
      }
      v4bf sv, mv;
      sv[0] = (__bf16)s.x; sv[1] = (__bf16)s.y; sv[2] = (__bf16)s.z; sv[3] = (__bf16)s.w;
      mv[0] = (__bf16)(ax * 0.0625f); mv[1] = (__bf16)(ay * 0.0625f);
      mv[2] = (__bf16)(az * 0.0625f); mv[3] = (__bf16)(aw * 0.0625f);
      *(v4bf*)(row + c)       = sv;
      *(v4bf*)(row + DIN + c) = mv;
    }
  } else {
    const __bf16* h_in = (const __bf16*)h_in_v;
    for (int c = sub * 8; c < DIN; c += 32) {
      v8bf s = *(const v8bf*)(h_in + (size_t)node * DIN + c);
      float acc[8];
#pragma unroll
      for (int i = 0; i < 8; ++i) acc[i] = 0.f;
#pragma unroll
      for (int f = 0; f < FANOUT_C; ++f) {
        v8bf t = *(const v8bf*)(h_in + (size_t)nb[f] * DIN + c);
#pragma unroll
        for (int i = 0; i < 8; ++i) acc[i] += (float)t[i];
      }
      v8bf m;
#pragma unroll
      for (int i = 0; i < 8; ++i) m[i] = (__bf16)(acc[i] * 0.0625f);
      *(v8bf*)(row + c)       = s;
      *(v8bf*)(row + DIN + c) = m;
    }
  }

  __syncthreads();

  // ---- Phase 2: WMMA GEMM: [32 x K2] @ [K2 x DOUT] ----
  constexpr int NT  = DOUT / 32;   // N-tiles per wave (wave owns half the columns)
  constexpr int KCH = K2 / 32;     // K chunks of 32
  const int wave   = tid >> 5;
  const int lane   = tid & 31;
  const int mIdx   = wave & 1;     // which 16-node M tile
  const int nHalf  = wave >> 1;    // which half of output columns
  const int laneLo = lane & 15;
  const int laneHi = lane >> 4;

  const unsigned char* aRow = &A_sh[(mIdx * 16 + laneLo) * RS];

  v8f acc[NT];
#pragma unroll
  for (int nt = 0; nt < NT; ++nt) acc[nt] = {};

  for (int k0 = 0; k0 < KCH; ++k0) {
    // A fragment (16-bit A 16x32 layout): low lanes K q0+q2, high lanes K q1+q3
    const int ka = k0 * 32 + laneHi * 8;
    AFrag a;
    a.q[0] = *(const uint4*)(aRow + (size_t)ka * 2);
    a.q[1] = *(const uint4*)(aRow + (size_t)(ka + 16) * 2);
    // B fragment (16-bit B 32x16 layout): lane = column N, low lanes K q0+q2, high lanes K q1+q3
    const int kb = k0 * 32 + laneHi * 8;
#pragma unroll
    for (int nt = 0; nt < NT; ++nt) {
      const int n = nHalf * (DOUT / 2) + nt * 16 + laneLo;
      const __bf16* wrow = Wbf + (size_t)n * K2 + kb;
      AFrag b;
      b.q[0] = *(const uint4*)(wrow);
      b.q[1] = *(const uint4*)(wrow + 16);
      acc[nt] = __builtin_amdgcn_wmma_f32_16x16x32_bf16(
          false, a.v, false, b.v, (short)0, acc[nt], false, false);
    }
  }

  // ---- Epilogue: ReLU + store ----
  const int gBase = blockIdx.x * NODES_PER_WG + mIdx * 16 + laneHi * 8;
#pragma unroll
  for (int nt = 0; nt < NT; ++nt) {
    const int col = nHalf * (DOUT / 2) + nt * 16 + laneLo;
#pragma unroll
    for (int v = 0; v < 8; ++v) {
      float x = acc[nt][v];
      x = x > 0.f ? x : 0.f;
      const size_t o = (size_t)(gBase + v) * DOUT + col;
      if constexpr (OUT_F32) ((float*)h_out_v)[o] = x;
      else                   ((__bf16*)h_out_v)[o] = (__bf16)x;
    }
  }
}

extern "C" void kernel_launch(void* const* d_in, const int* in_sizes, int n_in,
                              void* d_out, int out_size, void* d_ws, size_t ws_size,
                              hipStream_t stream) {
  const float* features = (const float*)d_in[0];
  const float* W1       = (const float*)d_in[1];
  const float* W2       = (const float*)d_in[2];
  const float* W3       = (const float*)d_in[3];
  const float* W4       = (const float*)d_in[4];
  const int*   neigh    = (const int*)d_in[6];
  const int N = in_sizes[5];  // nodes (arange) length

  constexpr int HID = 128, EMB = 64, DF = 256;
  constexpr int nW1 = HID * 2 * DF;   // 65536
  constexpr int nW2 = HID * 2 * HID;  // 32768
  constexpr int nW3 = HID * 2 * HID;  // 32768
  constexpr int nW4 = EMB * 2 * HID;  // 16384

  char* ws = (char*)d_ws;
  __bf16* W1b = (__bf16*)ws;
  __bf16* W2b = W1b + nW1;
  __bf16* W3b = W2b + nW2;
  __bf16* W4b = W3b + nW3;
  size_t off = (size_t)((char*)(W4b + nW4) - ws);
  off = (off + 255) & ~(size_t)255;
  __bf16* hA = (__bf16*)(ws + off);        // ping  (N*128 bf16 = 25.6 MB)
  __bf16* hB = hA + (size_t)N * HID;       // pong

  // Convert weights fp32 -> bf16 (L2-resident, reused by every workgroup)
  f32_to_bf16_kernel<<<(nW1 + 255) / 256, 256, 0, stream>>>(W1, W1b, nW1);
  f32_to_bf16_kernel<<<(nW2 + 255) / 256, 256, 0, stream>>>(W2, W2b, nW2);
  f32_to_bf16_kernel<<<(nW3 + 255) / 256, 256, 0, stream>>>(W3, W3b, nW3);
  f32_to_bf16_kernel<<<(nW4 + 255) / 256, 256, 0, stream>>>(W4, W4b, nW4);

  const int blocks = (N + NODES_PER_WG - 1) / NODES_PER_WG;  // 3125, exact
  sage_layer_kernel<DF,  HID, true,  false><<<blocks, THREADS, 0, stream>>>(features, neigh, W1b, hA);
  sage_layer_kernel<HID, HID, false, false><<<blocks, THREADS, 0, stream>>>(hA, neigh, W2b, hB);
  sage_layer_kernel<HID, HID, false, false><<<blocks, THREADS, 0, stream>>>(hB, neigh, W3b, hA);
  sage_layer_kernel<HID, EMB, false, true ><<<blocks, THREADS, 0, stream>>>(hA, neigh, W4b, d_out);
}
